// TypeInferModel_21268678050468
// MI455X (gfx1250) — compile-verified, matrix operand order checked by reading, not run on hardware
//
#include <hip/hip_runtime.h>
#include <hip/hip_bf16.h>

#define EPS   1e-5f
#define SLOPE 0.01f

typedef __attribute__((ext_vector_type(16))) __bf16 v16bf;
typedef __attribute__((ext_vector_type(8)))  float  v8f;

constexpr int GBM = 128, GBN = 128, GBK = 32, GLDT = GBK + 8;   // 80B padded LDS row stride

__device__ __forceinline__ unsigned short bf16u(float f) {
    unsigned int u = __builtin_bit_cast(unsigned int, f);
    u += 0x7fffu + ((u >> 16) & 1u);            // RNE to bf16
    return (unsigned short)(u >> 16);
}

#define ASYNC_B128(ldsaddr, gaddr) \
    asm volatile("global_load_async_to_lds_b128 %0, %1, off" \
                 :: "v"(ldsaddr), "v"(gaddr) : "memory")
#define WAIT_ASYNC() asm volatile("s_wait_asynccnt 0x0" ::: "memory")

// ---------------- one-time conversions ----------------
__global__ void k_f32_to_bf16(const float* __restrict__ in, unsigned short* __restrict__ out, int n) {
    int i = blockIdx.x * 256 + threadIdx.x;
    if (i < n) out[i] = bf16u(in[i]);
}

// W[K][N] fp32 -> Wt[N][K] bf16 (coalesced writes; strided reads are L2-hot)
template<int K, int N>
__global__ void k_wt_bf16(const float* __restrict__ W, unsigned short* __restrict__ Wt) {
    int idx = blockIdx.x * 256 + threadIdx.x;
    if (idx < K * N) {
        int n = idx / K, k = idx % K;
        Wt[idx] = bf16u(W[(size_t)k * N + n]);
    }
}

// ---------------- degree / normalization ----------------
__global__ void k_deg_init(float* __restrict__ deg, int n) {
    int i = blockIdx.x * 256 + threadIdx.x;
    if (i < n) deg[i] = 1.0f;                   // self-loop contribution
}

__global__ void k_deg_count(const long long* __restrict__ dst, float* __restrict__ deg, int e) {
    int i = blockIdx.x * 256 + threadIdx.x;
    if (i < e) atomicAdd(&deg[(int)dst[i]], 1.0f);
}

__global__ void k_deg_rsqrt(const float* __restrict__ deg, float* __restrict__ dinv, int n) {
    int i = blockIdx.x * 256 + threadIdx.x;
    if (i < n) dinv[i] = rsqrtf(deg[i]);
}

// ---------------- bf16 WMMA GEMM: C = act(A[M,K]bf16 @ Wt[N,K]bf16^T + bias) ----------------
// 256 threads = 8 waves; block tile 128x128; k-step 32; double-buffered LDS filled by
// GLOBAL_LOAD_ASYNC_TO_LDS_B128 (ASYNCcnt), WMMA of tile k overlaps DMA of tile k+1.
template<int N, int K, bool HAS_BIAS, bool LRELU>
__global__ __launch_bounds__(256)
void k_gemm_wmma(const __bf16* __restrict__ A, const __bf16* __restrict__ Wt,
                 const float* __restrict__ bias, float* __restrict__ C, int M)
{
    constexpr int NK = K / GBK;
    __shared__ __align__(16) __bf16 sA[2][GBM * GLDT];
    __shared__ __align__(16) __bf16 sB[2][GBN * GLDT];

    const int tid  = threadIdx.x;
    const int lane = tid & 31;
    const int wave = tid >> 5;
    const int wm = wave >> 1, wn = wave & 1;
    const int g  = lane >> 4, r  = lane & 15;
    const int blockM = blockIdx.x * GBM;
    const int blockN = blockIdx.y * GBN;

    // Async staging tasks: tile is 128 rows x 32 k (64B/row) = 512 x 16B chunks;
    // thread owns chunks (r0,c0) and (r1=r0+64,c0) for both A and B tiles.
    const int r0 = tid >> 2, c0 = tid & 3;
    const int r1 = r0 + 64;
    const int ga0 = (blockM + r0 < M) ? (blockM + r0) : (M - 1);   // clamp tail (stores guarded)
    const int ga1 = (blockM + r1 < M) ? (blockM + r1) : (M - 1);
    const __bf16* gA0 = A  + (size_t)ga0 * K + c0 * 8;
    const __bf16* gA1 = A  + (size_t)ga1 * K + c0 * 8;
    const __bf16* gB0 = Wt + (size_t)(blockN + r0) * K + c0 * 8;
    const __bf16* gB1 = Wt + (size_t)(blockN + r1) * K + c0 * 8;

    const unsigned ldsA = (unsigned)(size_t)&sA[0][0];   // LDS byte offset = addr[31:0]
    const unsigned ldsB = (unsigned)(size_t)&sB[0][0];
    const unsigned offA0 = (unsigned)(r0 * 80 + c0 * 16);
    const unsigned offA1 = (unsigned)(r1 * 80 + c0 * 16);
    constexpr unsigned BUFSZ = GBM * GLDT * 2;           // 10240B per buffer

    v8f acc[2][4] = {};

    // prologue: DMA tile 0
    {
        ASYNC_B128(ldsA + offA0, (unsigned long long)gA0);
        ASYNC_B128(ldsA + offA1, (unsigned long long)gA1);
        ASYNC_B128(ldsB + offA0, (unsigned long long)gB0);
        ASYNC_B128(ldsB + offA1, (unsigned long long)gB1);
    }
    WAIT_ASYNC();
    __syncthreads();

    #pragma unroll 2
    for (int kt = 0; kt < NK; ++kt) {
        const int cb = kt & 1;
        // kick off DMA for next k-tile into the alternate buffer (overlaps WMMA below)
        if (kt + 1 < NK) {
            const size_t ko = (size_t)(kt + 1) * GBK;
            const unsigned nb = (unsigned)((cb ^ 1) * BUFSZ);
            ASYNC_B128(ldsA + nb + offA0, (unsigned long long)(gA0 + ko));
            ASYNC_B128(ldsA + nb + offA1, (unsigned long long)(gA1 + ko));
            ASYNC_B128(ldsB + nb + offA0, (unsigned long long)(gB0 + ko));
            ASYNC_B128(ldsB + nb + offA1, (unsigned long long)(gB1 + ko));
        }
        // compute from current buffer
        #pragma unroll
        for (int mi = 0; mi < 2; ++mi) {
            const __bf16* pa = &sA[cb][(wm * 32 + mi * 16 + r) * GLDT];
            v16bf a;
            #pragma unroll
            for (int i = 0; i < 8; ++i) { a[i] = pa[g * 8 + i]; a[8 + i] = pa[16 + g * 8 + i]; }
            #pragma unroll
            for (int nj = 0; nj < 4; ++nj) {
                const __bf16* pb = &sB[cb][(wn * 64 + nj * 16 + r) * GLDT];
                v16bf b;
                #pragma unroll
                for (int i = 0; i < 8; ++i) { b[i] = pb[g * 8 + i]; b[8 + i] = pb[16 + g * 8 + i]; }
                acc[mi][nj] = __builtin_amdgcn_wmma_f32_16x16x32_bf16(
                    false, a, false, b, (short)0, acc[mi][nj], false, false);
            }
        }
        if (kt + 1 < NK) {
            WAIT_ASYNC();
            __syncthreads();
        }
    }

    // epilogue: C layout — M = reg + 8*g, N = lane%16; full blocks take guard-free path.
    const bool full = (blockM + GBM <= M);
    #pragma unroll
    for (int mi = 0; mi < 2; ++mi) {
        #pragma unroll
        for (int nj = 0; nj < 4; ++nj) {
            const int col  = blockN + wn * 64 + nj * 16 + r;
            const float bv = HAS_BIAS ? bias[col] : 0.0f;
            const int row0 = blockM + wm * 32 + mi * 16 + 8 * g;
            float* __restrict__ cp = C + (size_t)row0 * N + col;
            if (full) {
                #pragma unroll
                for (int rr = 0; rr < 8; ++rr) {
                    float v = acc[mi][nj][rr] + bv;
                    if (LRELU) v = (v >= 0.0f) ? v : SLOPE * v;
                    cp[rr * N] = v;                      // compile-time offsets
                }
            } else {
                #pragma unroll
                for (int rr = 0; rr < 8; ++rr) {
                    float v = acc[mi][nj][rr] + bv;
                    if (LRELU) v = (v >= 0.0f) ? v : SLOPE * v;
                    if (row0 + rr < M) cp[rr * N] = v;
                }
            }
        }
    }
}

// ---------------- GCN aggregation ----------------
__global__ void k_self_bias(const float* __restrict__ xw, const float* __restrict__ dinv,
                            const float* __restrict__ b, float* __restrict__ out, int n, int F) {
    int idx = blockIdx.x * 256 + threadIdx.x;
    if (idx < n * F) {
        int row = idx / F, c = idx % F;
        float dv = dinv[row];
        out[idx] = xw[idx] * dv * dv + b[c];
    }
}

__global__ void k_agg_edges(const long long* __restrict__ src, const long long* __restrict__ dst,
                            const float* __restrict__ dinv, const float* __restrict__ xw,
                            float* __restrict__ out, int E, int F) {
    int e = blockIdx.x * 256 + threadIdx.x;
    if (e >= E) return;
    int s = (int)src[e], d = (int)dst[e];
    float w = dinv[s] * dinv[d];
    int c = blockIdx.y * 4;
    const float4 v = *(const float4*)(xw + (size_t)s * F + c);
    float* o = out + (size_t)d * F + c;
    atomicAdd(o + 0, v.x * w);
    atomicAdd(o + 1, v.y * w);
    atomicAdd(o + 2, v.z * w);
    atomicAdd(o + 3, v.w * w);
}

// ---------------- BatchNorm ----------------
__global__ __launch_bounds__(256)
void k_bn_stats(const float* __restrict__ X, float* __restrict__ mean, float* __restrict__ rstd,
                int n, int F) {
    __shared__ float ssum[256], ssq[256];
    const int c = blockIdx.x;
    float s = 0.0f, q = 0.0f;
    for (int r = threadIdx.x; r < n; r += 256) {
        float v = X[(size_t)r * F + c];
        s += v; q += v * v;
    }
    ssum[threadIdx.x] = s; ssq[threadIdx.x] = q;
    __syncthreads();
    for (int st = 128; st > 0; st >>= 1) {
        if (threadIdx.x < st) {
            ssum[threadIdx.x] += ssum[threadIdx.x + st];
            ssq[threadIdx.x]  += ssq[threadIdx.x + st];
        }
        __syncthreads();
    }
    if (threadIdx.x == 0) {
        float m = ssum[0] / n;
        float v = ssq[0] / n - m * m;           // biased variance
        mean[c] = m;
        rstd[c] = rsqrtf(v + EPS);
    }
}

// BN + LeakyReLU, emitting bf16 (consumer is the next WMMA GEMM)
__global__ void k_bn_lrelu_bf16(const float* __restrict__ X, const float* __restrict__ mean,
                                const float* __restrict__ rstd, const float* __restrict__ g,
                                const float* __restrict__ be, unsigned short* __restrict__ out,
                                int n, int F) {
    int idx = blockIdx.x * 256 + threadIdx.x;
    if (idx < n * F) {
        int c = idx % F;
        float v = (X[idx] - mean[c]) * rstd[c] * g[c] + be[c];
        v = (v >= 0.0f) ? v : SLOPE * v;
        out[idx] = bf16u(v);
    }
}

// ---------------- head: logits = Y @ Wd2 + bd2 ; softmax ----------------
__global__ __launch_bounds__(256)
void k_head(const float* __restrict__ Y, const float* __restrict__ Wd2,
            const float* __restrict__ bd2, float* __restrict__ out, int n) {
    const int lane = threadIdx.x & 31;
    const int wave = threadIdx.x >> 5;
    const int row  = blockIdx.x * 8 + wave;
    if (row >= n) return;
    float acc[6] = {0.f, 0.f, 0.f, 0.f, 0.f, 0.f};
    const float* y = Y + (size_t)row * 4096;
    for (int k = lane; k < 4096; k += 32) {
        float x = y[k];
        const float* w = Wd2 + k * 6;
        #pragma unroll
        for (int j = 0; j < 6; ++j) acc[j] += x * w[j];
    }
    #pragma unroll
    for (int j = 0; j < 6; ++j) {
        #pragma unroll
        for (int m = 16; m > 0; m >>= 1) acc[j] += __shfl_xor(acc[j], m, 32);
    }
    if (lane == 0) {
        float mx = -1e30f;
        #pragma unroll
        for (int j = 0; j < 6; ++j) { acc[j] += bd2[j]; mx = fmaxf(mx, acc[j]); }
        float sum = 0.0f;
        #pragma unroll
        for (int j = 0; j < 6; ++j) { acc[j] = __expf(acc[j] - mx); sum += acc[j]; }
        float inv = 1.0f / sum;
        #pragma unroll
        for (int j = 0; j < 6; ++j) out[row * 6 + j] = acc[j] * inv;
    }
}

static inline int cdiv(int a, int b) { return (a + b - 1) / b; }

extern "C" void kernel_launch(void* const* d_in, const int* in_sizes, int n_in,
                              void* d_out, int out_size, void* d_ws, size_t ws_size,
                              hipStream_t stream) {
    const float*     x   = (const float*)d_in[0];
    const long long* ei  = (const long long*)d_in[1];   // int64 edge_index [2,E]
    const float*     W1  = (const float*)d_in[2];
    const float*     b1  = (const float*)d_in[3];
    const float*     W2  = (const float*)d_in[4];
    const float*     b2  = (const float*)d_in[5];
    const float*     g1  = (const float*)d_in[6];
    const float*     be1 = (const float*)d_in[7];
    const float*     g2  = (const float*)d_in[8];
    const float*     be2 = (const float*)d_in[9];
    const float*     Wd1 = (const float*)d_in[10];
    const float*     bd1 = (const float*)d_in[11];
    const float*     Wd2 = (const float*)d_in[12];
    const float*     bd2 = (const float*)d_in[13];

    const int N = in_sizes[0] / 128;
    const int E = in_sizes[1] / 2;
    const long long* src = ei;
    const long long* dst = ei + E;

    float* ws = (float*)d_ws;
    size_t off = 0;
    float* deg  = ws + off; off += (size_t)N;
    float* dinv = ws + off; off += (size_t)N;
    float* mean = ws + off; off += 512;
    float* rstd = ws + off; off += 512;
    float* bufA = ws + off; off += (size_t)N * 512;     // fp32: xw1 then xw2
    float* bufB = ws + off; off += (size_t)N * 512;     // fp32: h1 then h2 (pre-BN)
    float* bufC = ws + off; off += (size_t)N * 4096;    // fp32: y3
    unsigned short* xbf  = (unsigned short*)(ws + off); off += (size_t)N * 64;   // N*128 bf16
    unsigned short* h1bf = (unsigned short*)(ws + off); off += (size_t)N * 128;  // N*256 bf16
    unsigned short* h2bf = (unsigned short*)(ws + off); off += (size_t)N * 256;  // N*512 bf16
    unsigned short* W1t  = (unsigned short*)(ws + off); off += 16384;            // 128*256 bf16
    unsigned short* W2t  = (unsigned short*)(ws + off); off += 65536;            // 256*512 bf16
    unsigned short* Wd1t = (unsigned short*)(ws + off); off += 1048576;          // 512*4096 bf16

    // one-time conversions / transposes (bf16 operands for WMMA + async DMA)
    k_f32_to_bf16<<<cdiv(N * 128, 256), 256, 0, stream>>>(x, xbf, N * 128);
    k_wt_bf16<128, 256> <<<cdiv(128 * 256, 256), 256, 0, stream>>>(W1, W1t);
    k_wt_bf16<256, 512> <<<cdiv(256 * 512, 256), 256, 0, stream>>>(W2, W2t);
    k_wt_bf16<512, 4096><<<cdiv(512 * 4096, 256), 256, 0, stream>>>(Wd1, Wd1t);

    // normalization D^{-1/2}
    k_deg_init <<<cdiv(N, 256), 256, 0, stream>>>(deg, N);
    k_deg_count<<<cdiv(E, 256), 256, 0, stream>>>(dst, deg, E);
    k_deg_rsqrt<<<cdiv(N, 256), 256, 0, stream>>>(deg, dinv, N);

    // ---- layer 1: GCNConv(128->256) + BN + LeakyReLU ----
    k_gemm_wmma<256, 128, false, false><<<dim3(cdiv(N, 128), 256 / 128), 256, 0, stream>>>(
        (const __bf16*)xbf, (const __bf16*)W1t, nullptr, bufA, N);
    k_self_bias<<<cdiv(N * 256, 256), 256, 0, stream>>>(bufA, dinv, b1, bufB, N, 256);
    k_agg_edges<<<dim3(cdiv(E, 256), 256 / 4), 256, 0, stream>>>(src, dst, dinv, bufA, bufB, E, 256);
    k_bn_stats <<<256, 256, 0, stream>>>(bufB, mean, rstd, N, 256);
    k_bn_lrelu_bf16<<<cdiv(N * 256, 256), 256, 0, stream>>>(bufB, mean, rstd, g1, be1, h1bf, N, 256);

    // ---- layer 2: GCNConv(256->512) + BN + LeakyReLU ----
    k_gemm_wmma<512, 256, false, false><<<dim3(cdiv(N, 128), 512 / 128), 256, 0, stream>>>(
        (const __bf16*)h1bf, (const __bf16*)W2t, nullptr, bufA, N);
    k_self_bias<<<cdiv(N * 512, 256), 256, 0, stream>>>(bufA, dinv, b2, bufB, N, 512);
    k_agg_edges<<<dim3(cdiv(E, 256), 512 / 4), 256, 0, stream>>>(src, dst, dinv, bufA, bufB, E, 512);
    k_bn_stats <<<512, 256, 0, stream>>>(bufB, mean, rstd, N, 512);
    k_bn_lrelu_bf16<<<cdiv(N * 512, 256), 256, 0, stream>>>(bufB, mean, rstd, g2, be2, h2bf, N, 512);

    // ---- dense1 (512->4096) + bias + LeakyReLU, the FLOP-dominant GEMM ----
    k_gemm_wmma<4096, 512, true, true><<<dim3(cdiv(N, 128), 4096 / 128), 256, 0, stream>>>(
        (const __bf16*)h2bf, (const __bf16*)Wd1t, bd1, bufC, N);

    // ---- dense2 (4096->6) + softmax ----
    k_head<<<cdiv(N, 8), 256, 0, stream>>>(bufC, Wd2, bd2, (float*)d_out, N);
}
